// MultiheadA_27650999452430
// MI455X (gfx1250) — matrix operand order repaired and checked by reading, not compiled, not run on hardware
//
#include <hip/hip_runtime.h>
#include <hip/hip_bf16.h>
#include <math.h>
#include <stdint.h>

typedef __attribute__((ext_vector_type(16))) _Float16 v16h;
typedef __attribute__((ext_vector_type(8)))  _Float16 v8h;
typedef __attribute__((ext_vector_type(8)))  float    v8f;
typedef __attribute__((ext_vector_type(4)))  float    v4f;

#define BB    2
#define TT    1500
#define DD    1024
#define HH    16
#define DHH   64
#define TP    1504        // T padded to 16
#define ROWS  3000        // B*T
#define ROWSP 3008        // rows padded to 16
#define QT    94          // 16-row tiles over T
#define QT2   47          // 32-row blocks over T
#define RT2   94          // 32-row blocks over ROWSP
#define SCALEQ 0.35355339059327373f  // 64^-0.25

// ---------------------------------------------------------------- utilities
__global__ void zero_kernel(uint32_t* p, size_t n) {
    size_t i  = (size_t)blockIdx.x * blockDim.x + threadIdx.x;
    size_t st = (size_t)gridDim.x * blockDim.x;
    for (; i < n; i += st) p[i] = 0u;
}

// one block: f0 mean -> mel freqs -> cos/sin tables (scaled by f0[t])
__global__ void prep_kernel(const float* __restrict__ f0, const float* __restrict__ theta,
                            float* __restrict__ ctab, float* __restrict__ stab) {
    __shared__ float red[256];
    __shared__ float freq[32];
    int tid = threadIdx.x;
    float s = 0.f;
    for (int i = tid; i < TT; i += 256) s += f0[i];
    red[tid] = s; __syncthreads();
    for (int w = 128; w > 0; w >>= 1) { if (tid < w) red[tid] += red[tid + w]; __syncthreads(); }
    if (tid < 32) {
        float melmax = 2595.0f * log10f(1.0f + 8000.0f / 700.0f);
        float lin    = melmax * (float)tid / 31.0f;
        float te     = red[0] / (float)TT + theta[0];
        freq[tid]    = te / 220.0f * 700.0f * (powf(10.0f, lin / 2595.0f) - 1.0f) / 1000.0f;
    }
    __syncthreads();
    for (int idx = tid; idx < TT * 32; idx += 256) {
        int t = idx >> 5, i = idx & 31;
        float ang = (float)t * freq[i];
        float r   = f0[t];
        ctab[idx] = r * cosf(ang);
        stab[idx] = r * sinf(ang);
    }
}

__global__ void cvt_x_kernel(const float* __restrict__ x, _Float16* __restrict__ xh) {
    size_t i = (size_t)blockIdx.x * blockDim.x + threadIdx.x;
    if (i < (size_t)ROWS * DD) xh[i] = (_Float16)x[i];
}

// f32 weights -> f16 transposed (N x K, K contiguous) for B-operand loads
__global__ void cvt_wt_kernel(const float* __restrict__ wq, const float* __restrict__ wk,
                              const float* __restrict__ wv, const float* __restrict__ wo,
                              _Float16* __restrict__ wt) {
    size_t i = (size_t)blockIdx.x * blockDim.x + threadIdx.x;
    if (i >= (size_t)4 * DD * DD) return;
    int w = (int)(i / (DD * DD));
    int rem = (int)(i % (DD * DD));
    int k = rem / DD, n = rem % DD;
    const float* src = (w == 0) ? wq : (w == 1) ? wk : (w == 2) ? wv : wo;
    wt[(size_t)w * DD * DD + (size_t)n * DD + k] = (_Float16)src[rem];
}

__device__ __forceinline__ v16h combine16(v8h lo, v8h hi) {
    v16h r;
    #pragma unroll
    for (int e = 0; e < 8; e++) { r[e] = lo[e]; r[8 + e] = hi[e]; }
    return r;
}

// ---------------------------------------------------------------- QKV GEMM (32x32 per wave)
__global__ __launch_bounds__(32)
void qkv_kernel(const _Float16* __restrict__ xh, const _Float16* __restrict__ wt,
                const float* __restrict__ bq, const float* __restrict__ bv,
                float* __restrict__ qf, float* __restrict__ kf, _Float16* __restrict__ vt) {
    int l = threadIdx.x;
    int rt = blockIdx.x, ctile = blockIdx.y, mat = blockIdx.z;
    const _Float16* w = wt + (size_t)mat * DD * DD;
    int kloA = (l < 16) ? 0 : 8;    // A layout: lanes 16-31 hold K+8 / K+24
    int kloB = (l < 16) ? 0 : 8;    // B layout: lanes 16-31 hold K+8 / K+24
    const _Float16* ap0 = xh + (size_t)(rt * 32 + (l & 15)) * DD;
    const _Float16* ap1 = ap0 + (size_t)16 * DD;
    const _Float16* bp0 = w + (size_t)(ctile * 32 + (l & 15)) * DD;
    const _Float16* bp1 = bp0 + (size_t)16 * DD;
    v8f z = {0.f,0.f,0.f,0.f,0.f,0.f,0.f,0.f};
    v8f acc[2][2] = {{z, z}, {z, z}};
    for (int kc = 0; kc < DD; kc += 32) {
        v16h A0 = combine16(*(const v8h*)(ap0 + kc + kloA), *(const v8h*)(ap0 + kc + 16 + kloA));
        v16h A1 = combine16(*(const v8h*)(ap1 + kc + kloA), *(const v8h*)(ap1 + kc + 16 + kloA));
        v16h B0 = combine16(*(const v8h*)(bp0 + kc + kloB), *(const v8h*)(bp0 + kc + kloB + 16));
        v16h B1 = combine16(*(const v8h*)(bp1 + kc + kloB), *(const v8h*)(bp1 + kc + kloB + 16));
        acc[0][0] = __builtin_amdgcn_wmma_f32_16x16x32_f16(false, A0, false, B0, (short)0, acc[0][0], false, false);
        acc[0][1] = __builtin_amdgcn_wmma_f32_16x16x32_f16(false, A0, false, B1, (short)0, acc[0][1], false, false);
        acc[1][0] = __builtin_amdgcn_wmma_f32_16x16x32_f16(false, A1, false, B0, (short)0, acc[1][0], false, false);
        acc[1][1] = __builtin_amdgcn_wmma_f32_16x16x32_f16(false, A1, false, B1, (short)0, acc[1][1], false, false);
    }
    #pragma unroll
    for (int mt = 0; mt < 2; mt++) {
        #pragma unroll
        for (int nt = 0; nt < 2; nt++) {
            int n = ctile * 32 + nt * 16 + (l & 15);
            #pragma unroll
            for (int r = 0; r < 8; r++) {
                int grow = rt * 32 + mt * 16 + r + ((l < 16) ? 0 : 8);
                if (grow >= ROWS) continue;
                float val = acc[mt][nt][r];
                if (mat == 0) {
                    qf[(size_t)grow * DD + n] = val + bq[n];
                } else if (mat == 1) {
                    kf[(size_t)grow * DD + n] = val;
                } else {
                    val += bv[n];
                    int b = grow / TT, t = grow % TT;
                    int h = n >> 6, dh = n & 63;
                    vt[((size_t)(b * HH + h) * DHH + dh) * TP + t] = (_Float16)val;
                }
            }
        }
    }
}

// ---------------------------------------------------------------- RoPE + scale + zscale
__global__ void rope_kernel(const float* __restrict__ qf, const float* __restrict__ kf,
                            const float* __restrict__ ctab, const float* __restrict__ stab,
                            _Float16* __restrict__ qh, _Float16* __restrict__ kh,
                            float* __restrict__ zs) {
    size_t idx = (size_t)blockIdx.x * blockDim.x + threadIdx.x;
    if (idx >= (size_t)BB * HH * TP * 32) return;
    int i = (int)(idx & 31);
    size_t tmp = idx >> 5;
    int t  = (int)(tmp % TP);
    int bh = (int)(tmp / TP);
    size_t qo = (((size_t)bh * TP) + t) * 64 + 2 * i;
    if (t < TT) {
        int b = bh / HH, h = bh % HH;
        size_t so = ((size_t)(b * TT + t)) * DD + h * 64 + 2 * i;
        float c = ctab[t * 32 + i], s = stab[t * 32 + i];
        float qr = qf[so], qi = qf[so + 1];
        qh[qo]     = (_Float16)((qr * c - qi * s) * SCALEQ);
        qh[qo + 1] = (_Float16)((qr * s + qi * c) * SCALEQ);
        float kr = kf[so], ki = kf[so + 1];
        float k0 = kr * c - ki * s;
        kh[qo]     = (_Float16)(k0 * SCALEQ);
        kh[qo + 1] = (_Float16)((kr * s + ki * c) * SCALEQ);
        if (i == 0) {
            float fz = fminf(fmaxf(log1pf(expf(1e-4f)), 1e-6f), 1e-3f);
            zs[(size_t)bh * TP + t] = (k0 == 0.0f) ? fz : 1.0f;
        }
    } else {
        qh[qo] = (_Float16)0.f; qh[qo + 1] = (_Float16)0.f;
        kh[qo] = (_Float16)0.f; kh[qo + 1] = (_Float16)0.f;
        if (i == 0) zs[(size_t)bh * TP + t] = 1.0f;
    }
}

// ---------------------------------------------------------------- QK^T (2 q-tiles per wave; writes masked qk)
__global__ __launch_bounds__(32)
void qk_kernel(const _Float16* __restrict__ qh, const _Float16* __restrict__ kh,
               const float* __restrict__ zs, float* __restrict__ qkout) {
    int l = threadIdx.x;
    int qt = blockIdx.x, bh = blockIdx.y;
    int kloA = (l < 16) ? 0 : 8;
    int kloB = (l < 16) ? 0 : 8;
    int qrow0 = qt * 32 + (l & 15);
    const _Float16* qp0 = qh + ((size_t)bh * TP + qrow0) * 64;
    const _Float16* qp1 = qp0 + (size_t)16 * 64;
    // A tiles (held in registers for the entire k loop): [mtile][khalf]
    v16h A00 = combine16(*(const v8h*)(qp0 + kloA),      *(const v8h*)(qp0 + 16 + kloA));
    v16h A01 = combine16(*(const v8h*)(qp0 + 32 + kloA), *(const v8h*)(qp0 + 48 + kloA));
    v16h A10 = combine16(*(const v8h*)(qp1 + kloA),      *(const v8h*)(qp1 + 16 + kloA));
    v16h A11 = combine16(*(const v8h*)(qp1 + 32 + kloA), *(const v8h*)(qp1 + 48 + kloA));
    for (int kt = 0; kt < QT; kt++) {
        int kcol = kt * 16 + (l & 15);
        const _Float16* kp = kh + ((size_t)bh * TP + kcol) * 64;
        v16h B0 = combine16(*(const v8h*)(kp + kloB),      *(const v8h*)(kp + kloB + 16));
        v16h B1 = combine16(*(const v8h*)(kp + 32 + kloB), *(const v8h*)(kp + 32 + kloB + 16));
        v8f z = {0.f,0.f,0.f,0.f,0.f,0.f,0.f,0.f};
        v8f acc0 = z, acc1 = z;
        acc0 = __builtin_amdgcn_wmma_f32_16x16x32_f16(false, A00, false, B0, (short)0, acc0, false, false);
        acc0 = __builtin_amdgcn_wmma_f32_16x16x32_f16(false, A01, false, B1, (short)0, acc0, false, false);
        acc1 = __builtin_amdgcn_wmma_f32_16x16x32_f16(false, A10, false, B0, (short)0, acc1, false, false);
        acc1 = __builtin_amdgcn_wmma_f32_16x16x32_f16(false, A11, false, B1, (short)0, acc1, false, false);
        float zc = zs[(size_t)bh * TP + kcol];
        #pragma unroll
        for (int mt = 0; mt < 2; mt++) {
            #pragma unroll
            for (int r = 0; r < 8; r++) {
                int q = qt * 32 + mt * 16 + r + ((l < 16) ? 0 : 8);
                if (q < TT && kcol < TT) {
                    float v = (mt == 0) ? acc0[r] : acc1[r];
                    float mval = (kcol > q) ? -1000000000.0f : 0.0f;
                    v = (v + mval * zc) * zc;
                    qkout[((size_t)bh * TT + q) * TT + kcol] = v;
                }
            }
        }
    }
}

// ---------------------------------------------------------------- per-row softmax stats
__global__ void stat_kernel(const float* __restrict__ qkout,
                            float* __restrict__ rmax, float* __restrict__ rsum) {
    int lane = threadIdx.x & 31;
    int wid  = (int)(((size_t)blockIdx.x * blockDim.x + threadIdx.x) >> 5);
    if (wid >= BB * HH * TT) return;
    const float* rowp = qkout + (size_t)wid * TT;
    float m = -3.4e38f;
    for (int c = lane; c < TT; c += 32) m = fmaxf(m, rowp[c]);
    for (int off = 16; off > 0; off >>= 1) m = fmaxf(m, __shfl_xor(m, off, 32));
    float s = 0.f;
    for (int c = lane; c < TT; c += 32) s += expf(rowp[c] - m);
    for (int off = 16; off > 0; off >>= 1) s += __shfl_xor(s, off, 32);
    if (lane == 0) { rmax[wid] = m; rsum[wid] = s; }
}

// ---------------------------------------------------------------- P @ V (re-reads qk, wmma vs v^T)
__global__ __launch_bounds__(32)
void pv_kernel(const float* __restrict__ qkout, const float* __restrict__ rmax,
               const float* __restrict__ rsum, const _Float16* __restrict__ vt,
               _Float16* __restrict__ ah) {
    int l = threadIdx.x;
    int qt = blockIdx.x, bh = blockIdx.y;
    int m = l & 15;
    int q = qt * 16 + m;
    int qc = (q < TT) ? q : (TT - 1);
    float rm = rmax[(size_t)bh * TT + qc];
    const float* rowp = qkout + ((size_t)bh * TT + qc) * TT;
    int kloA = (l < 16) ? 0 : 8;
    int kloB = (l < 16) ? 0 : 8;
    v8f zero = {0.f,0.f,0.f,0.f,0.f,0.f,0.f,0.f};
    v8f acc[4];
    #pragma unroll
    for (int i = 0; i < 4; i++) acc[i] = zero;
    const _Float16* vbase = vt + (size_t)bh * DHH * TP;
    for (int kc = 0; kc < TP; kc += 32) {
        v16h A;
        if (kc + 31 < TT) {                       // fast path: fully in-bounds row chunk
            v4f u0 = *(const v4f*)(rowp + kc + kloA);
            v4f u1 = *(const v4f*)(rowp + kc + kloA + 4);
            v4f u2 = *(const v4f*)(rowp + kc + 16 + kloA);
            v4f u3 = *(const v4f*)(rowp + kc + 16 + kloA + 4);
            #pragma unroll
            for (int e = 0; e < 4; e++) {
                A[e]      = (_Float16)expf(u0[e] - rm);
                A[4 + e]  = (_Float16)expf(u1[e] - rm);
                A[8 + e]  = (_Float16)expf(u2[e] - rm);
                A[12 + e] = (_Float16)expf(u3[e] - rm);
            }
        } else {                                  // guarded tail (k >= 1500 -> p = 0)
            #pragma unroll
            for (int e = 0; e < 8; e++) {
                int k1 = kc + kloA + e;
                int k2 = kc + 16 + kloA + e;
                float x1 = (k1 < TT) ? rowp[k1] : -3.4e38f;
                float x2 = (k2 < TT) ? rowp[k2] : -3.4e38f;
                A[e]     = (_Float16)expf(x1 - rm);
                A[8 + e] = (_Float16)expf(x2 - rm);
            }
        }
        #pragma unroll
        for (int nt = 0; nt < 4; nt++) {
            int dh = nt * 16 + (l & 15);
            const _Float16* vp = vbase + (size_t)dh * TP + kc + kloB;
            v16h Bm = combine16(*(const v8h*)vp, *(const v8h*)(vp + 16));
            acc[nt] = __builtin_amdgcn_wmma_f32_16x16x32_f16(false, A, false, Bm, (short)0, acc[nt], false, false);
        }
    }
    int b = bh / HH, h = bh % HH;
    #pragma unroll
    for (int r = 0; r < 8; r++) {
        int mm = r + ((l < 16) ? 0 : 8);
        int qq = qt * 16 + mm;
        if (qq >= TT) continue;
        float inv = 1.0f / rsum[(size_t)bh * TT + qq];
        #pragma unroll
        for (int nt = 0; nt < 4; nt++) {
            int n = nt * 16 + (l & 15);
            ah[((size_t)(b * TT + qq)) * DD + h * 64 + n] = (_Float16)(acc[nt][r] * inv);
        }
    }
}

// ---------------------------------------------------------------- output projection (32x32 per wave)
__global__ __launch_bounds__(32)
void outproj_kernel(const _Float16* __restrict__ ah, const _Float16* __restrict__ wot,
                    const float* __restrict__ bo, float* __restrict__ out) {
    int l = threadIdx.x;
    int rt = blockIdx.x, ctile = blockIdx.y;
    int kloA = (l < 16) ? 0 : 8;
    int kloB = (l < 16) ? 0 : 8;
    const _Float16* ap0 = ah + (size_t)(rt * 32 + (l & 15)) * DD;
    const _Float16* ap1 = ap0 + (size_t)16 * DD;
    const _Float16* bp0 = wot + (size_t)(ctile * 32 + (l & 15)) * DD;
    const _Float16* bp1 = bp0 + (size_t)16 * DD;
    v8f z = {0.f,0.f,0.f,0.f,0.f,0.f,0.f,0.f};
    v8f acc[2][2] = {{z, z}, {z, z}};
    for (int kc = 0; kc < DD; kc += 32) {
        v16h A0 = combine16(*(const v8h*)(ap0 + kc + kloA), *(const v8h*)(ap0 + kc + 16 + kloA));
        v16h A1 = combine16(*(const v8h*)(ap1 + kc + kloA), *(const v8h*)(ap1 + kc + 16 + kloA));
        v16h B0 = combine16(*(const v8h*)(bp0 + kc + kloB), *(const v8h*)(bp0 + kc + kloB + 16));
        v16h B1 = combine16(*(const v8h*)(bp1 + kc + kloB), *(const v8h*)(bp1 + kc + kloB + 16));
        acc[0][0] = __builtin_amdgcn_wmma_f32_16x16x32_f16(false, A0, false, B0, (short)0, acc[0][0], false, false);
        acc[0][1] = __builtin_amdgcn_wmma_f32_16x16x32_f16(false, A0, false, B1, (short)0, acc[0][1], false, false);
        acc[1][0] = __builtin_amdgcn_wmma_f32_16x16x32_f16(false, A1, false, B0, (short)0, acc[1][0], false, false);
        acc[1][1] = __builtin_amdgcn_wmma_f32_16x16x32_f16(false, A1, false, B1, (short)0, acc[1][1], false, false);
    }
    #pragma unroll
    for (int mt = 0; mt < 2; mt++) {
        #pragma unroll
        for (int nt = 0; nt < 2; nt++) {
            int n = ctile * 32 + nt * 16 + (l & 15);
            #pragma unroll
            for (int r = 0; r < 8; r++) {
                int grow = rt * 32 + mt * 16 + r + ((l < 16) ? 0 : 8);
                if (grow >= ROWS) continue;
                out[(size_t)grow * DD + n] = acc[mt][nt][r] + bo[n];
            }
        }
    }
}

// ---------------------------------------------------------------- launcher
extern "C" void kernel_launch(void* const* d_in, const int* in_sizes, int n_in,
                              void* d_out, int out_size, void* d_ws, size_t ws_size,
                              hipStream_t stream) {
    const float* x     = (const float*)d_in[0];
    const float* f0    = (const float*)d_in[1];
    // d_in[2] = mask (causal + -1e9), reproduced analytically
    const float* Wq    = (const float*)d_in[3];
    const float* bq    = (const float*)d_in[4];
    const float* Wk    = (const float*)d_in[5];
    const float* Wv    = (const float*)d_in[6];
    const float* bv    = (const float*)d_in[7];
    const float* Wo    = (const float*)d_in[8];
    const float* bo    = (const float*)d_in[9];
    const float* theta = (const float*)d_in[10];

    char* ws = (char*)d_ws;
    size_t o = 0;
    auto alloc = [&](size_t b) { size_t r = o; o = (o + b + 255) & ~(size_t)255; return r; };
    // zero-initialized (padded) region first, contiguous:
    size_t xh_o = alloc((size_t)ROWSP * DD * 2);
    size_t qh_o = alloc((size_t)BB * HH * TP * DHH * 2);
    size_t kh_o = alloc((size_t)BB * HH * TP * DHH * 2);
    size_t vt_o = alloc((size_t)BB * HH * DHH * TP * 2);
    size_t ah_o = alloc((size_t)ROWSP * DD * 2);
    size_t zeroEnd = o;
    size_t wt_o = alloc((size_t)4 * DD * DD * 2);
    size_t qf_o = alloc((size_t)ROWS * DD * 4);
    size_t kf_o = alloc((size_t)ROWS * DD * 4);
    size_t ct_o = alloc((size_t)TT * 32 * 4);
    size_t st_o = alloc((size_t)TT * 32 * 4);
    size_t zs_o = alloc((size_t)BB * HH * TP * 4);
    size_t rm_o = alloc((size_t)BB * HH * TT * 4);
    size_t rs_o = alloc((size_t)BB * HH * TT * 4);

    _Float16* xh = (_Float16*)(ws + xh_o);
    _Float16* qh = (_Float16*)(ws + qh_o);
    _Float16* kh = (_Float16*)(ws + kh_o);
    _Float16* vt = (_Float16*)(ws + vt_o);
    _Float16* ah = (_Float16*)(ws + ah_o);
    _Float16* wt = (_Float16*)(ws + wt_o);
    float* qf = (float*)(ws + qf_o);
    float* kf = (float*)(ws + kf_o);
    float* ct = (float*)(ws + ct_o);
    float* st = (float*)(ws + st_o);
    float* zs = (float*)(ws + zs_o);
    float* rm = (float*)(ws + rm_o);
    float* rs = (float*)(ws + rs_o);

    float* out   = (float*)d_out;
    float* qkout = out + (size_t)BB * TT * DD;

    zero_kernel<<<2048, 256, 0, stream>>>((uint32_t*)ws, zeroEnd / 4);
    prep_kernel<<<1, 256, 0, stream>>>(f0, theta, ct, st);
    cvt_x_kernel<<<((size_t)ROWS * DD + 255) / 256, 256, 0, stream>>>(x, xh);
    cvt_wt_kernel<<<((size_t)4 * DD * DD + 255) / 256, 256, 0, stream>>>(Wq, Wk, Wv, Wo, wt);
    qkv_kernel<<<dim3(RT2, DD / 32, 3), 32, 0, stream>>>(xh, wt, bq, bv, qf, kf, vt);
    rope_kernel<<<((size_t)BB * HH * TP * 32 + 255) / 256, 256, 0, stream>>>(qf, kf, ct, st, qh, kh, zs);
    qk_kernel<<<dim3(QT2, BB * HH), 32, 0, stream>>>(qh, kh, zs, qkout);
    stat_kernel<<<((size_t)BB * HH * TT * 32 + 255) / 256, 256, 0, stream>>>(qkout, rm, rs);
    pv_kernel<<<dim3(QT, BB * HH), 32, 0, stream>>>(qkout, rm, rs, vt, ah);
    outproj_kernel<<<dim3(RT2, DD / 32), 32, 0, stream>>>(ah, wt + (size_t)3 * DD * DD, bo, out);
}
